// AttentionPool_3547642986619
// MI455X (gfx1250) — matrix operand order repaired and checked by reading, not compiled, not run on hardware
//
#include <hip/hip_runtime.h>
#include <hip/hip_bf16.h>

// ---------------------------------------------------------------------------
// AttentionPool for MI455X (gfx1250, wave32, WMMA)
//   scores = tanh(x@W1+b1)@W2+b2 ; softmax over sorted segments ; out = seg_sum(w*x)
//
// Roofline: x (512MB fp32) must be streamed twice (score pass + pooling pass)
// -> ~45us at 23.3 TB/s. The 34.4 GFLOP score matmul is done in bf16 WMMA
// (~15us at bf16 matrix rates) so it hides under the HBM stream.
// ---------------------------------------------------------------------------

typedef __attribute__((ext_vector_type(16))) __bf16 v16bf;
typedef __attribute__((ext_vector_type(8)))  float  v8f;

#define DD 512          // feature dim
#define HH 128          // hidden dim (D/4)
#define WPB 8           // waves per block in score kernel
#define RPW 16          // rows per wave (one WMMA M tile)

__device__ __forceinline__ __bf16 f2bf(float f) {
    unsigned u = __builtin_bit_cast(unsigned, f);
    u += 0x7FFFu + ((u >> 16) & 1u);                 // round-to-nearest-even
    return __builtin_bit_cast(__bf16, (unsigned short)(u >> 16));
}

// ---------------------------------------------------------------------------
// Kernel 0: convert W1 -> bf16 scratch, zero out[] and denom[]
// ---------------------------------------------------------------------------
__global__ void init_kernel(const float* __restrict__ W1, __bf16* __restrict__ w1bf,
                            float* __restrict__ out, float* __restrict__ denom,
                            int w1n, int outn, int dn) {
    int i = blockIdx.x * blockDim.x + threadIdx.x;
    if (i < w1n) w1bf[i] = f2bf(W1[i]);
    if (i < outn) out[i] = 0.0f;
    if (i < dn) denom[i] = 0.0f;
}

// ---------------------------------------------------------------------------
// Kernel 1: scores via bf16 WMMA.  Block = 256 thr = 8 wave32.
// Wave w computes rows [base, base+16): h = x_tile(16x512) @ W1(512x128),
// then score[m] = sum_n tanh(h[m][n]+b1[n])*W2[n] + b2.
// W1 K-slices staged transposed in double-buffered LDS.
// Pipelining: A-tile global loads for step s+1 issued during step s's WMMA
// burst; all 8 B fragments preloaded into distinct VGPRs so the ds_loads
// batch behind a single s_wait_dscnt and the 8 WMMAs issue back-to-back.
// ---------------------------------------------------------------------------
__global__ __launch_bounds__(256) void score_kernel(
    const float* __restrict__ x, const __bf16* __restrict__ w1bf,
    const float* __restrict__ b1, const float* __restrict__ W2,
    const float* __restrict__ b2, float* __restrict__ scores,
    float* __restrict__ partial_max) {

    __shared__ __bf16 ldsBT[2][HH * 32];   // [buf][n*32 + k'], 8KB each
    __shared__ float  wave_max[WPB];

    const int tid  = threadIdx.x;
    const int lane = tid & 31;
    const int wv   = tid >> 5;
    const int half = lane >> 4;            // 0: lanes 0-15, 1: lanes 16-31
    const int l16  = lane & 15;

    const int row_base = (blockIdx.x * WPB + wv) * RPW;
    const float* xr = x + (size_t)(row_base + l16) * DD;

    v8f acc[8];
#pragma unroll
    for (int t = 0; t < 8; ++t) acc[t] = (v8f)(0.0f);

    // A fragment addressing (contiguous K order, matching B):
    //   lane half h, row M=l16: VGPR0-7 hold K = kk + 16h + {0..15}
    float4 a0 = *(const float4*)(xr + half * 16);
    float4 a1 = *(const float4*)(xr + half * 16 + 4);
    float4 a2 = *(const float4*)(xr + half * 16 + 8);
    float4 a3 = *(const float4*)(xr + half * 16 + 12);

    for (int step = 0; step < DD / 32; ++step) {
        const int kk  = step * 32;
        const int buf = step & 1;
        // stage W1[kk..kk+31][0..127] transposed: ldsBT[buf][n*32+k'] (coalesced reads)
#pragma unroll
        for (int i = 0; i < 16; ++i) {
            int j  = tid + i * 256;        // 0..4095
            int kp = j >> 7;               // 0..31
            int n  = j & 127;
            ldsBT[buf][n * 32 + kp] = w1bf[(size_t)(kk + kp) * HH + n];
        }
        __syncthreads();   // staged buffer visible; prior buffer's readers done

        // convert current A tile to bf16 (RNE); co-executes with XDL pipe
        v16bf af;
        af[0]=f2bf(a0.x); af[1]=f2bf(a0.y); af[2]=f2bf(a0.z);  af[3]=f2bf(a0.w);
        af[4]=f2bf(a1.x); af[5]=f2bf(a1.y); af[6]=f2bf(a1.z);  af[7]=f2bf(a1.w);
        af[8]=f2bf(a2.x); af[9]=f2bf(a2.y); af[10]=f2bf(a2.z); af[11]=f2bf(a2.w);
        af[12]=f2bf(a3.x);af[13]=f2bf(a3.y);af[14]=f2bf(a3.z); af[15]=f2bf(a3.w);

        // software-pipeline: issue next step's A-tile loads now (latency
        // overlaps with the B loads + WMMA burst below)
        if (step < DD / 32 - 1) {
            const int kn = kk + 32;
            a0 = *(const float4*)(xr + kn + half * 16);
            a1 = *(const float4*)(xr + kn + half * 16 + 4);
            a2 = *(const float4*)(xr + kn + half * 16 + 8);
            a3 = *(const float4*)(xr + kn + half * 16 + 12);
        }

        // B fragment in contiguous K order (same as A): lane = column N,
        // half 0 holds K=kk+0..15, half 1 holds K=kk+16..31 (packed pairs) ->
        // contiguous 32B per lane from the transposed LDS tile.
        // Preload ALL 8 fragments into distinct VGPRs so the 16 ds_load_b128
        // batch behind one wait and the WMMAs issue back-to-back.
        v16bf bf[8];
#pragma unroll
        for (int t = 0; t < 8; ++t)
            bf[t] = *(const v16bf*)&ldsBT[buf][(t * 16 + l16) * 32 + half * 16];

#pragma unroll
        for (int t = 0; t < 8; ++t)
            acc[t] = __builtin_amdgcn_wmma_f32_16x16x32_bf16(
                false, af, false, bf[t], (short)0, acc[t], false, false);
    }

    // Epilogue: C layout -> VGPR r holds row M=r (lanes 0-15) / M=r+8 (16-31),
    // column N = 16t + l16.  score[m] = sum_n tanh(h+b1)*W2 + b2.
    float rowsum[8];
#pragma unroll
    for (int r = 0; r < 8; ++r) rowsum[r] = 0.0f;
#pragma unroll
    for (int t = 0; t < 8; ++t) {
        int n = t * 16 + l16;
        float b1v = b1[n];
        float w2v = W2[n];
#pragma unroll
        for (int r = 0; r < 8; ++r)
            rowsum[r] += tanhf(acc[t][r] + b1v) * w2v;
    }
    // butterfly over the 16 lanes of each half (xor<16 stays in-half on wave32)
#pragma unroll
    for (int m = 1; m <= 8; m <<= 1)
#pragma unroll
        for (int r = 0; r < 8; ++r)
            rowsum[r] += __shfl_xor(rowsum[r], m, 32);

    const float bias2 = b2[0];
    if (l16 == 0) {
#pragma unroll
        for (int r = 0; r < 8; ++r)
            scores[row_base + half * 8 + r] = rowsum[r] + bias2;
    }
    // block max for numerically-stable softmax
    float mx = -3.402823466e38f;
#pragma unroll
    for (int r = 0; r < 8; ++r) mx = fmaxf(mx, rowsum[r] + bias2);
    mx = fmaxf(mx, __shfl_xor(mx, 16, 32));
    if (lane == 0) wave_max[wv] = mx;
    __syncthreads();
    if (tid == 0) {
        float bm = wave_max[0];
#pragma unroll
        for (int w = 1; w < WPB; ++w) bm = fmaxf(bm, wave_max[w]);
        partial_max[blockIdx.x] = bm;
    }
}

// ---------------------------------------------------------------------------
// Kernel 2: single-block max reduction of per-block partial maxes
// ---------------------------------------------------------------------------
__global__ __launch_bounds__(256) void reduce_max_kernel(
    const float* __restrict__ partial, int n, float* __restrict__ gmax) {
    __shared__ float wm[8];
    int tid = threadIdx.x;
    float m = -3.402823466e38f;
    for (int i = tid; i < n; i += 256) m = fmaxf(m, partial[i]);
#pragma unroll
    for (int s = 1; s < 32; s <<= 1) m = fmaxf(m, __shfl_xor(m, s, 32));
    if ((tid & 31) == 0) wm[tid >> 5] = m;
    __syncthreads();
    if (tid == 0) {
        float bm = wm[0];
#pragma unroll
        for (int w = 1; w < 8; ++w) bm = fmaxf(bm, wm[w]);
        *gmax = bm;
    }
}

// ---------------------------------------------------------------------------
// Kernel 3: e = exp(s - gmax) in-place; denom[batch] += e (sorted, low contention)
// ---------------------------------------------------------------------------
__global__ __launch_bounds__(256) void exp_denom_kernel(
    float* __restrict__ scores, const long long* __restrict__ batch,
    const float* __restrict__ gmax, float* __restrict__ denom, int n) {
    int i = blockIdx.x * blockDim.x + threadIdx.x;
    if (i >= n) return;
    float e = __expf(scores[i] - *gmax);
    scores[i] = e;
    atomicAdd(&denom[(int)batch[i]], e);
}

// ---------------------------------------------------------------------------
// Kernel 4: out[b] += (e/denom[b]) * x[row].  512 thr = 1 column each,
// 256 sorted rows per block; register accumulation, atomic flush only on
// segment boundary (~2 flushes/block for avg 256-row segments).
// ---------------------------------------------------------------------------
__global__ __launch_bounds__(512) void pool_kernel(
    const float* __restrict__ x, const float* __restrict__ e,
    const long long* __restrict__ batch, const float* __restrict__ denom,
    float* __restrict__ out) {
    __shared__ float wbuf[256];
    __shared__ int   bbuf[256];
    const int tid = threadIdx.x;
    const int r0  = blockIdx.x * 256;
    if (tid < 256) {
        int b = (int)batch[r0 + tid];
        bbuf[tid] = b;
        wbuf[tid] = e[r0 + tid] / (denom[b] + 1e-8f);
    }
    __syncthreads();

    const int d = tid;                 // 0..511
    float acc = 0.0f;
    int cur = bbuf[0];
    for (int i = 0; i < 256; ++i) {
        if ((i & 7) == 0 && i + 8 < 256)
            __builtin_prefetch(&x[(size_t)(r0 + i + 8) * DD + d], 0, 1);
        int b = bbuf[i];
        if (b != cur) {
            atomicAdd(&out[(size_t)cur * DD + d], acc);
            acc = 0.0f;
            cur = b;
        }
        acc = fmaf(wbuf[i], x[(size_t)(r0 + i) * DD + d], acc);
    }
    atomicAdd(&out[(size_t)cur * DD + d], acc);
}

// ---------------------------------------------------------------------------
// Host launcher
// ---------------------------------------------------------------------------
extern "C" void kernel_launch(void* const* d_in, const int* in_sizes, int n_in,
                              void* d_out, int out_size, void* d_ws, size_t ws_size,
                              hipStream_t stream) {
    const float*     x     = (const float*)d_in[0];
    const long long* batch = (const long long*)d_in[1];
    const float*     W1    = (const float*)d_in[2];
    const float*     b1    = (const float*)d_in[3];
    const float*     W2    = (const float*)d_in[4];
    const float*     b2    = (const float*)d_in[5];
    float*           out   = (float*)d_out;

    const int N = in_sizes[1];                 // rows (262144)
    const int B = out_size / DD;               // segments (1024)
    const int w1n = DD * HH;                   // 65536

    // workspace layout
    char*  ws    = (char*)d_ws;
    __bf16* w1bf = (__bf16*)ws;                                  // 128 KB
    float* e     = (float*)(ws + (size_t)w1n * sizeof(__bf16));  // N floats (scores -> exp)
    const int nblk_score = N / (WPB * RPW);                      // 2048
    float* pmax  = e + N;
    float* gmax  = pmax + nblk_score;
    float* denom = gmax + 1;
    (void)ws_size; (void)n_in;

    int init_n = out_size > w1n ? out_size : w1n;
    init_kernel<<<(init_n + 255) / 256, 256, 0, stream>>>(W1, w1bf, out, denom,
                                                          w1n, out_size, B);
    score_kernel<<<nblk_score, 256, 0, stream>>>(x, w1bf, b1, W2, b2, e, pmax);
    reduce_max_kernel<<<1, 256, 0, stream>>>(pmax, nblk_score, gmax);
    exp_denom_kernel<<<(N + 255) / 256, 256, 0, stream>>>(e, batch, gmax, denom, N);
    pool_kernel<<<N / 256, 512, 0, stream>>>(x, e, batch, denom, out);
}
